// QuantLinear_45896020525851
// MI455X (gfx1250) — matrix operand order repaired and checked by reading, not compiled, not run on hardware
//
#include <hip/hip_runtime.h>

#define M_DIM 4096
#define K_DIM 4096
#define N_DIM 12288
#define GS    128
#define NG    (K_DIM / GS)   // 32 groups
#define NW    (N_DIM / 8)    // int32 words per weight row = 1536

typedef __attribute__((ext_vector_type(16))) __bf16 bf16x16;
typedef __attribute__((ext_vector_type(2)))  __bf16 bf16x2;
typedef __attribute__((ext_vector_type(8)))  float  f32x8;

#define LDSA_STRIDE 40   // 32 k + 8 pad (bf16 elems); keeps 16B alignment, staggers banks
#define LDSB_STRIDE 40

struct U32x8 { uint4 a, b; };   // 32 bytes == bf16x16

__device__ __forceinline__ unsigned int pack_bf16(float lo, float hi) {
  bf16x2 p = { (__bf16)lo, (__bf16)hi };   // -> single v_cvt_pk_bf16_f32
  return __builtin_bit_cast(unsigned int, p);
}

__global__ __launch_bounds__(256)
void qlinear_wmma_kernel(const float* __restrict__ x,
                         const int*   __restrict__ qweight,
                         const int*   __restrict__ qzeros,
                         const float* __restrict__ scales,
                         const float* __restrict__ bias,
                         float*       __restrict__ out) {
  __shared__ unsigned short sA[128 * LDSA_STRIDE]; // A tile, [row][k], bf16
  __shared__ unsigned short sB[128 * LDSB_STRIDE]; // B tile transposed, [n][k], bf16

  const int tid  = threadIdx.x;
  const int lane = tid & 31;       // wave32
  const int wave = tid >> 5;       // 0..7
  const int wm   = wave >> 1;      // 0..3 : which 32-row slab
  const int wn   = wave & 1;       // 0..1 : which 64-col slab

  const int n0 = blockIdx.x * 128;
  const int m0 = blockIdx.y * 128;

  // fixed dequant responsibility per thread: one word-column, adjacent k pair
  const int w    = tid & 15;           // word within block's 16-word span
  const int kr   = tid >> 4;           // 0..15 -> k rows (2kr, 2kr+1)
  const int wcol = (n0 >> 3) + w;      // absolute qweight word column

  f32x8 acc[2][4] = {};

  for (int g = 0; g < NG; ++g) {
    // ---- per-group zeros + scales for this thread's 8 output columns ----
    const int zw = qzeros[g * NW + wcol];
    int   zq[8];
    float sc[8];
#pragma unroll
    for (int j = 0; j < 8; ++j) {
      zq[j] = (zw >> (4 * j)) & 0xF;
      sc[j] = scales[(size_t)g * N_DIM + n0 + w * 8 + j];
    }

    for (int kk = 0; kk < GS / 32; ++kk) {
      const int k0 = g * GS + kk * 32;

      __syncthreads();  // previous iteration's fragment reads complete

      // ---- stage A: 128x32 fp32 -> bf16 LDS ----
#pragma unroll
      for (int i = 0; i < 4; ++i) {
        const int f   = tid + 256 * i;   // float4 index 0..1023
        const int row = f >> 3;
        const int col = (f & 7) * 4;
        const float4 v = *reinterpret_cast<const float4*>(
            x + (size_t)(m0 + row) * K_DIM + k0 + col);
        uint2 u;
        u.x = pack_bf16(v.x, v.y);
        u.y = pack_bf16(v.z, v.w);
        *reinterpret_cast<uint2*>(&sA[row * LDSA_STRIDE + col]) = u;
      }
      if (k0 + 32 < K_DIM) {  // uniform branch; hint next A tile into cache
        __builtin_prefetch(x + (size_t)(m0 + (tid >> 1)) * K_DIM + k0 + 32, 0, 1);
      }

      // ---- stage B: dequant 32x128 int4 -> bf16, transposed [n][k] ----
      const int q0 = qweight[(size_t)(k0 + 2 * kr + 0) * NW + wcol];
      const int q1 = qweight[(size_t)(k0 + 2 * kr + 1) * NW + wcol];
#pragma unroll
      for (int j = 0; j < 8; ++j) {
        const int nl = w * 8 + j;
        const float f0 = (float)(((q0 >> (4 * j)) & 0xF) - zq[j]) * sc[j];
        const float f1 = (float)(((q1 >> (4 * j)) & 0xF) - zq[j]) * sc[j];
        *reinterpret_cast<unsigned int*>(&sB[nl * LDSB_STRIDE + 2 * kr]) =
            pack_bf16(f0, f1);   // adjacent k pair -> one b32 store
      }

      __syncthreads();

      // ---- A fragments (ISA 16-bit A 16x32 layout) ----
      const int arow   = wm * 32 + (lane & 15);
      const int kabase = (lane >> 4) * 8;          // lanes 0-15: K0..7/16..23; 16-31: +8
      bf16x16 afrag[2];
#pragma unroll
      for (int tm = 0; tm < 2; ++tm) {
        const unsigned short* pa = &sA[(arow + tm * 16) * LDSA_STRIDE];
        U32x8 t;
        t.a = *reinterpret_cast<const uint4*>(pa + kabase);
        t.b = *reinterpret_cast<const uint4*>(pa + kabase + 16);
        afrag[tm] = __builtin_bit_cast(bf16x16, t);
      }

      // ---- B fragments: lane = column, lanes 0-15 K0..7/16..23, 16-31 K8..15/24..31 ----
      const int bkbase = (lane >> 4) * 8;
      bf16x16 bfrag[4];
#pragma unroll
      for (int tn = 0; tn < 4; ++tn) {
        const unsigned short* pb = &sB[(wn * 64 + tn * 16 + (lane & 15)) * LDSB_STRIDE];
        U32x8 t;
        t.a = *reinterpret_cast<const uint4*>(pb + bkbase);
        t.b = *reinterpret_cast<const uint4*>(pb + bkbase + 16);
        bfrag[tn] = __builtin_bit_cast(bf16x16, t);
      }

      // ---- 8 WMMAs per wave per K-step ----
#pragma unroll
      for (int tm = 0; tm < 2; ++tm)
#pragma unroll
        for (int tn = 0; tn < 4; ++tn)
          acc[tm][tn] = __builtin_amdgcn_wmma_f32_16x16x32_bf16(
              false, afrag[tm], false, bfrag[tn],
              (short)0, acc[tm][tn], false, false);
    }
  }

  // ---- epilogue: bias + fp32 store (each lane's 8 elems share one column) ----
#pragma unroll
  for (int tm = 0; tm < 2; ++tm) {
    const int row0 = m0 + wm * 32 + tm * 16 + (lane >> 4) * 8;
#pragma unroll
    for (int tn = 0; tn < 4; ++tn) {
      const int col = n0 + wn * 64 + tn * 16 + (lane & 15);
      const float bv = bias[col];
#pragma unroll
      for (int e = 0; e < 8; ++e) {
        out[(size_t)(row0 + e) * N_DIM + col] = acc[tm][tn][e] + bv;
      }
    }
  }
}

extern "C" void kernel_launch(void* const* d_in, const int* in_sizes, int n_in,
                              void* d_out, int out_size, void* d_ws, size_t ws_size,
                              hipStream_t stream) {
  const float* x       = (const float*)d_in[0];
  const int*   qweight = (const int*)d_in[1];
  const int*   qzeros  = (const int*)d_in[2];
  const float* scales  = (const float*)d_in[3];
  const float* bias    = (const float*)d_in[4];
  float* out = (float*)d_out;
  (void)in_sizes; (void)n_in; (void)out_size; (void)d_ws; (void)ws_size;

  dim3 grid(N_DIM / 128, M_DIM / 128);  // 96 x 32 blocks
  qlinear_wmma_kernel<<<grid, 256, 0, stream>>>(x, qweight, qzeros, scales, bias, out);
}
